// ScaledHead_1949915152920
// MI455X (gfx1250) — matrix operand order repaired and checked by reading, not compiled, not run on hardware
//
#include <hip/hip_runtime.h>
#include <hip/hip_bf16.h>
#include <math.h>

typedef _Float16 h16;
typedef __attribute__((ext_vector_type(16))) _Float16 v16h;
typedef __attribute__((ext_vector_type(8)))  float    v8f;
typedef unsigned int u32x4 __attribute__((ext_vector_type(4)));
typedef int          i32x8 __attribute__((ext_vector_type(8)));
typedef int          i32x4 __attribute__((ext_vector_type(4)));

#define B_   8
#define T_   2048
#define E_   512
#define D_   1024
#define NTOK (B_ * T_)
#define LDS_PITCH 34   /* 32 f16 + 1-dword TDM pad -> 17 dwords/row, coprime with 64 banks */

union FragU { v16h v; unsigned u[8]; };

// A fragment: 16x32 f16 (M x K), row m = lane&15.
// lanes 0-15 hold K = {0..7, 16..23}; lanes 16-31 hold K = {8..15, 24..31}; 2 f16/VGPR.
__device__ __forceinline__ v16h load_a_frag(const h16* __restrict__ A, int lda,
                                            int m0, int k0, int lane) {
  FragU f;
  const h16* p = A + (size_t)(m0 + (lane & 15)) * lda + (k0 + ((lane >> 4) << 3));
#pragma unroll
  for (int j = 0; j < 4; ++j) f.u[j]     = *(const unsigned*)(p + 2 * j);
#pragma unroll
  for (int j = 0; j < 4; ++j) f.u[4 + j] = *(const unsigned*)(p + 16 + 2 * j);
  return f.v;
}

// B fragment from N-major storage Bt[n*ldb + k] (k contiguous).
// lanes 0-15: col n, K = {0..7, 16..23}; lanes 16-31: col n, K = {8..15, 24..31}.
__device__ __forceinline__ v16h load_b_frag(const h16* __restrict__ Bt, int ldb,
                                            int n0, int k0, int lane) {
  FragU f;
  const h16* p = Bt + (size_t)(n0 + (lane & 15)) * ldb + (k0 + ((lane >> 4) << 3));
#pragma unroll
  for (int j = 0; j < 4; ++j) f.u[j]     = *(const unsigned*)(p + 2 * j);
#pragma unroll
  for (int j = 0; j < 4; ++j) f.u[4 + j] = *(const unsigned*)(p + 16 + 2 * j);
  return f.v;
}

// C/D layout: lane n (0-15) VGPR r -> (M=r, N=n); lane n+16 VGPR r -> (M=r+8, N=n).
__device__ __forceinline__ void store_c_f16(h16* __restrict__ C, int ldc,
                                            int m0, int n0, int lane, v8f c) {
  int n  = n0 + (lane & 15);
  int mb = m0 + ((lane >> 4) << 3);
#pragma unroll
  for (int r = 0; r < 8; ++r) C[(size_t)(mb + r) * ldc + n] = (h16)c[r];
}

__device__ __forceinline__ void atomicMaxFloat(float* addr, float val) {
  unsigned* ua = (unsigned*)addr;
  unsigned old = *ua;
  for (;;) {
    float oldf = __uint_as_float(old);
    if (oldf >= val) break;
    unsigned assumed = old;
    old = atomicCAS(ua, assumed, __float_as_uint(val));
    if (old == assumed) break;
  }
}

// ---- Tensor Data Mover: async 2D tile load Global -> LDS (rows x 32 f16) ----
// D# per CDNA5 ISA. data_size=2B; LDS pad: +1 dword per
// 16 dwords (one 32-f16 row) -> LDS row pitch = 34 f16. Tracked by TENSORcnt.
__device__ __forceinline__ void tdm_load_2d(const void* gptr, unsigned lds_off,
                                            int rows, int stride_elems) {
  unsigned long long ga = (unsigned long long)(uintptr_t)gptr;
  u32x4 g0;
  g0[0] = 1u;                                              // count=1, user mode
  g0[1] = lds_off;                                         // lds_addr (bytes)
  g0[2] = (unsigned)ga;                                    // global_addr[31:0]
  g0[3] = (unsigned)((ga >> 32) & 0x01ffffffu) | (2u << 30); // addr[56:32] | type=2
  i32x8 g1;
  g1[0] = (int)((1u << 16) | (1u << 20) | (3u << 22));     // data_size=2B, pad_en, ivl=16dw, amt=1dw
  g1[1] = 0;                                               // tensor_dim0 lo16 = 0
  g1[2] = 0x4000;                                          // tensor_dim0 = 0x40000000 (no OOB)
  g1[3] = (int)(0x4000u | (32u << 16));                    // tensor_dim1 hi | tile_dim0=32
  g1[4] = rows;                                            // tile_dim1 = rows, tile_dim2=0
  g1[5] = stride_elems;                                    // tensor_dim0_stride lo32
  g1[6] = 0;
  g1[7] = 0;
  i32x4 z4 = {0, 0, 0, 0};
#if defined(__clang_major__) && (__clang_major__ >= 23)
  i32x8 z8 = {0, 0, 0, 0, 0, 0, 0, 0};
  __builtin_amdgcn_tensor_load_to_lds(g0, g1, z4, z4, z8, 0);
#else
  __builtin_amdgcn_tensor_load_to_lds(g0, g1, z4, z4, 0);
#endif
}

// ---------------- conversion / init kernels ----------------

__global__ void k_cast_f16(const float* __restrict__ in, h16* __restrict__ out, int n) {
  int i = blockIdx.x * blockDim.x + threadIdx.x;
  if (i < n) out[i] = (h16)in[i];
}

// out[n*R + k] = in[k*C + n]   (k in [0,R), n in [0,C))
__global__ void k_transpose_cast(const float* __restrict__ in, h16* __restrict__ out,
                                 int R, int C) {
  int i = blockIdx.x * blockDim.x + threadIdx.x;
  if (i >= R * C) return;
  int k = i % R, n = i / R;
  out[(size_t)n * R + k] = (h16)in[(size_t)k * C + n];
}

__global__ void k_init_stats(float* __restrict__ colmax, float* __restrict__ denom, int n) {
  int i = blockIdx.x * blockDim.x + threadIdx.x;
  if (i < n) { colmax[i] = -3.0e38f; denom[i] = 0.0f; }
}

// ------- TDM + LDS double-buffered WMMA GEMM: 128x64 block tile, 16x64/wave -------

__global__ void k_gemm_f16_lds(const h16* __restrict__ Ab, const h16* __restrict__ Bb,
                               h16* __restrict__ Cb, int K, int lda, int ldb, int ldc,
                               size_t sA, size_t sB, size_t sC) {
  __shared__ h16 shA[2][128 * LDS_PITCH];
  __shared__ h16 shB[2][64 * LDS_PITCH];
  const h16* A = Ab + sA * blockIdx.z;
  const h16* B = Bb + sB * blockIdx.z;
  h16*       C = Cb + sC * blockIdx.z;
  int lane = threadIdx.x & 31, wave = threadIdx.x >> 5;
  int mblk = blockIdx.x * 128;
  int n0   = blockIdx.y * 64;
  bool issuer = (threadIdx.x < 32);   // wave 0 issues TDM ops (one per wave; EXEC ignored)
  unsigned ldsA[2] = { (unsigned)(uintptr_t)&shA[0][0], (unsigned)(uintptr_t)&shA[1][0] };
  unsigned ldsB[2] = { (unsigned)(uintptr_t)&shB[0][0], (unsigned)(uintptr_t)&shB[1][0] };

  if (issuer) {
    tdm_load_2d(A + (size_t)mblk * lda, ldsA[0], 128, lda);
    tdm_load_2d(B + (size_t)n0 * ldb,   ldsB[0], 64,  ldb);
  }
  v8f acc[4] = {{}, {}, {}, {}};
  int nsteps = K / 32;
  for (int i = 0; i < nsteps; ++i) {
    if (i + 1 < nsteps) {
      if (issuer) {  // prefetch next k-tile into the other buffer
        int k1 = (i + 1) * 32;
        tdm_load_2d(A + (size_t)mblk * lda + k1, ldsA[(i + 1) & 1], 128, lda);
        tdm_load_2d(B + (size_t)n0 * ldb + k1,   ldsB[(i + 1) & 1], 64,  ldb);
      }
      __builtin_amdgcn_s_wait_tensorcnt(2);  // TDM completes in order: stage i done
    } else {
      __builtin_amdgcn_s_wait_tensorcnt(0);
    }
    __syncthreads();
    const h16* sAp = &shA[i & 1][0];
    const h16* sBp = &shB[i & 1][0];
    v16h a = load_a_frag(sAp, LDS_PITCH, wave * 16, 0, lane);
#pragma unroll
    for (int sub = 0; sub < 4; ++sub) {
      v16h b = load_b_frag(sBp, LDS_PITCH, sub * 16, 0, lane);
      acc[sub] = __builtin_amdgcn_wmma_f32_16x16x32_f16(false, a, false, b, (short)0,
                                                        acc[sub], false, false);
    }
    __syncthreads();  // everyone done reading buf (i&1) before it is refilled
  }
  int m0 = mblk + wave * 16;
  store_c_f16(C, ldc, m0, n0 +  0, lane, acc[0]);
  store_c_f16(C, ldc, m0, n0 + 16, lane, acc[1]);
  store_c_f16(C, ldc, m0, n0 + 32, lane, acc[2]);
  store_c_f16(C, ldc, m0, n0 + 48, lane, acc[3]);
}

// ---------------- scores: S = sqrt(C) * K @ Q^T, causal mask, col-max ----------------

__global__ void k_scores(const h16* __restrict__ Kb, const h16* __restrict__ Qb,
                         h16* __restrict__ Sb, float* __restrict__ colmaxb) {
  const float scale = 22.62741699796952f;  // sqrt(512)
  const h16* Kmat = Kb + (size_t)T_ * D_ * blockIdx.z;
  const h16* Qmat = Qb + (size_t)T_ * D_ * blockIdx.z;
  h16*       S    = Sb + (size_t)T_ * T_ * blockIdx.z;
  float* colmax   = colmaxb + (size_t)T_ * blockIdx.z;
  int lane = threadIdx.x & 31, wave = threadIdx.x >> 5;
  int m0 = (blockIdx.x * 8 + wave) * 16;  // t tile
  int n0 = blockIdx.y * 64;               // s tile
  int nlane = lane & 15;
  int mb = ((lane >> 4) << 3);
  const h16 ninf = (h16)(-__builtin_huge_valf());

  if (n0 > m0 + 15) {  // fully masked 16x64 region
#pragma unroll
    for (int sub = 0; sub < 4; ++sub)
#pragma unroll
      for (int r = 0; r < 8; ++r)
        S[(size_t)(m0 + mb + r) * T_ + n0 + sub * 16 + nlane] = ninf;
    return;
  }

  v8f acc[4] = {{}, {}, {}, {}};
  for (int k0 = 0; k0 < D_; k0 += 32) {
    __builtin_prefetch(Kmat + (size_t)(m0 + (lane & 15)) * D_ + k0 + 64, 0, 1);
    v16h a = load_a_frag(Kmat, D_, m0, k0, lane);
#pragma unroll
    for (int sub = 0; sub < 4; ++sub) {
      v16h b = load_b_frag(Qmat, D_, n0 + sub * 16, k0, lane);
      acc[sub] = __builtin_amdgcn_wmma_f32_16x16x32_f16(false, a, false, b, (short)0,
                                                        acc[sub], false, false);
    }
  }
#pragma unroll
  for (int sub = 0; sub < 4; ++sub) {
    int s = n0 + sub * 16 + nlane;
    float lmax = -3.0e38f;
#pragma unroll
    for (int r = 0; r < 8; ++r) {
      int t = m0 + mb + r;
      float val = acc[sub][r] * scale;
      bool valid = (s <= t);
      S[(size_t)t * T_ + s] = valid ? (h16)val : ninf;
      if (valid) lmax = fmaxf(lmax, val);
    }
    if (lmax > -2.9e38f) atomicMaxFloat(&colmax[s], lmax);
  }
}

// ---------------- columnwise exp + denom ----------------

__global__ void k_softmax_col(h16* __restrict__ Pt, const float* __restrict__ colmax,
                              float* __restrict__ denom) {
  const int CH = T_ / 32;  // 64 chunks of 32 t-values
  int i = blockIdx.x * blockDim.x + threadIdx.x;
  int c = i % CH, s = (i / CH) % T_, b = i / (CH * T_);
  size_t base = (size_t)b * T_ * T_;
  float m = colmax[b * T_ + s];
  float sum = 0.0f;
  for (int j = 0; j < 32; ++j) {
    int t = c * 32 + j;
    size_t off = base + (size_t)t * T_ + s;
    float p = __expf((float)Pt[off] - m);  // masked entries: exp(-inf)=0
    Pt[off] = (h16)p;
    sum += p;
  }
  atomicAdd(&denom[b * T_ + s], sum);
}

// ---------------- V' = V/denom, transposed to [D][T] per batch ----------------

__global__ void k_vnorm_t(const h16* __restrict__ V, const float* __restrict__ denom,
                          h16* __restrict__ Vt) {
  int i = blockIdx.x * blockDim.x + threadIdx.x;
  int s = i % T_, d = (i / T_) % D_, b = i / (T_ * D_);
  float val = (float)V[((size_t)b * T_ + s) * D_ + d] / denom[b * T_ + s];
  Vt[(size_t)b * D_ * T_ + (size_t)d * T_ + s] = (h16)val;
}

// ---------------- final: relu(out2 @ Wl + bl), f32 out ----------------

__global__ void k_gemm_final(const h16* __restrict__ A, const h16* __restrict__ B,
                             const float* __restrict__ bias, float* __restrict__ C) {
  int lane = threadIdx.x & 31, wave = threadIdx.x >> 5;
  int m0 = (blockIdx.x * 8 + wave) * 16;
  int n0 = blockIdx.y * 64;
  v8f acc[4] = {{}, {}, {}, {}};
  for (int k0 = 0; k0 < D_; k0 += 32) {
    __builtin_prefetch(A + (size_t)(m0 + (lane & 15)) * D_ + k0 + 64, 0, 1);
    v16h a = load_a_frag(A, D_, m0, k0, lane);
#pragma unroll
    for (int sub = 0; sub < 4; ++sub) {
      v16h b = load_b_frag(B, D_, n0 + sub * 16, k0, lane);
      acc[sub] = __builtin_amdgcn_wmma_f32_16x16x32_f16(false, a, false, b, (short)0,
                                                        acc[sub], false, false);
    }
  }
  int nlane = lane & 15;
  int mb = m0 + ((lane >> 4) << 3);
#pragma unroll
  for (int sub = 0; sub < 4; ++sub) {
    int n = n0 + sub * 16 + nlane;
    float bv = bias[n];
#pragma unroll
    for (int r = 0; r < 8; ++r)
      C[(size_t)(mb + r) * E_ + n] = fmaxf(acc[sub][r] + bv, 0.0f);
  }
}

// ---------------- host ----------------

extern "C" void kernel_launch(void* const* d_in, const int* in_sizes, int n_in,
                              void* d_out, int out_size, void* d_ws, size_t ws_size,
                              hipStream_t stream) {
  (void)in_sizes; (void)n_in; (void)out_size; (void)ws_size;
  const float* x  = (const float*)d_in[0];
  const float* Wk = (const float*)d_in[1];
  const float* Wq = (const float*)d_in[2];
  const float* Wv = (const float*)d_in[3];
  const float* Wl = (const float*)d_in[4];
  const float* bl = (const float*)d_in[5];
  float* out = (float*)d_out;

  char* w = (char*)d_ws;
  auto alloc = [&](size_t bytes) -> char* {
    char* p = w; w += (bytes + 255) & ~(size_t)255; return p;
  };
  h16*   x16   = (h16*)  alloc((size_t)NTOK * E_ * 2);
  h16*   Wkqvt = (h16*)  alloc((size_t)3 * D_ * E_ * 2);     // Wk^T,Wq^T,Wv^T contiguous
  h16*   Wlt   = (h16*)  alloc((size_t)E_ * D_ * 2);
  h16*   KQV   = (h16*)  alloc((size_t)3 * NTOK * D_ * 2);   // K,Q,V contiguous
  h16*   Pt    = (h16*)  alloc((size_t)B_ * T_ * T_ * 2);    // scores then P, [t][s]
  h16*   Vt    = (h16*)  alloc((size_t)B_ * D_ * T_ * 2);    // V'/denom transposed
  h16*   out2  = (h16*)  alloc((size_t)NTOK * D_ * 2);
  float* colmax= (float*)alloc((size_t)B_ * T_ * 4);
  float* denom = (float*)alloc((size_t)B_ * T_ * 4);
  h16* Kbuf = KQV;
  h16* Qbuf = KQV + (size_t)NTOK * D_;
  h16* Vbuf = KQV + (size_t)2 * NTOK * D_;

  // 1) casts / transposes / stat init
  k_cast_f16<<<(NTOK * E_) / 256, 256, 0, stream>>>(x, x16, NTOK * E_);
  k_transpose_cast<<<(E_ * D_) / 256, 256, 0, stream>>>(Wk, Wkqvt,                       E_, D_);
  k_transpose_cast<<<(E_ * D_) / 256, 256, 0, stream>>>(Wq, Wkqvt + (size_t)D_ * E_,     E_, D_);
  k_transpose_cast<<<(E_ * D_) / 256, 256, 0, stream>>>(Wv, Wkqvt + (size_t)2 * D_ * E_, E_, D_);
  k_transpose_cast<<<(D_ * E_) / 256, 256, 0, stream>>>(Wl, Wlt, D_, E_);
  k_init_stats<<<(B_ * T_) / 256, 256, 0, stream>>>(colmax, denom, B_ * T_);

  // 2) fused QKV projection (z = 3 selects weight/output), TDM+LDS pipeline
  k_gemm_f16_lds<<<dim3(NTOK / 128, D_ / 64, 3), 256, 0, stream>>>(
      x16, Wkqvt, KQV, E_, E_, E_, D_,
      (size_t)0, (size_t)D_ * E_, (size_t)NTOK * D_);

  // 3) scores + mask + column max (z = batch)
  k_scores<<<dim3(T_ / 128, T_ / 64, B_), 256, 0, stream>>>(Kbuf, Qbuf, Pt, colmax);

  // 4) column softmax numerator + denom
  k_softmax_col<<<(B_ * T_ * (T_ / 32)) / 256, 256, 0, stream>>>(Pt, colmax, denom);

  // 5) normalize V by denom, transpose to [D][T]
  k_vnorm_t<<<(B_ * D_ * T_) / 256, 256, 0, stream>>>(Vbuf, denom, Vt);

  // 6) out2 = P @ V' per batch, TDM+LDS pipeline
  k_gemm_f16_lds<<<dim3(T_ / 128, D_ / 64, B_), 256, 0, stream>>>(
      Pt, Vt, out2, T_, T_, T_, D_,
      (size_t)T_ * T_, (size_t)D_ * T_, (size_t)T_ * D_);

  // 7) final projection + bias + relu
  k_gemm_final<<<dim3(NTOK / 128, E_ / 64, 1), 256, 0, stream>>>(out2, Wlt, bl, out);
}
